// MambaFeature_64355789964070
// MI455X (gfx1250) — matrix operand order repaired and checked by reading, not compiled, not run on hardware
//
#include <hip/hip_runtime.h>
#include <hip/hip_bf16.h>
#include <stdint.h>

// ---------------------------------------------------------------------------
// Types for CDNA5 WMMA (wave32, v_wmma_f32_16x16x32_f16)
// ---------------------------------------------------------------------------
typedef __attribute__((ext_vector_type(16))) _Float16 v16h;
typedef __attribute__((ext_vector_type(8)))  _Float16 v8h;
typedef __attribute__((ext_vector_type(8)))  float    v8f;

#define CAT16(lo, hi) __builtin_shufflevector(lo, hi, 0,1,2,3,4,5,6,7,8,9,10,11,12,13,14,15)

__device__ __forceinline__ v8h zero_v8h() {
    v8h v;
#pragma unroll
    for (int e = 0; e < 8; ++e) v[e] = (_Float16)0.f;
    return v;
}

__device__ __forceinline__ float softplusf(float x) {
    if (x > 20.f) return x;
    return __logf(1.f + __expf(x));
}
__device__ __forceinline__ float siluf(float x) {
    return x / (1.f + __expf(-x));
}

// ---------------------------------------------------------------------------
// Model dimensions
// ---------------------------------------------------------------------------
#define BATCH 512
#define NB    200
#define PP    144      // 12*12
#define LSEQ  64
#define DM    144
#define DI    288
#define DS    16
#define DC    4
#define DR    9

// ---------------------------------------------------------------------------
// f32 -> f16 copy
// ---------------------------------------------------------------------------
__global__ void cvt_f16_kernel(const float* __restrict__ s, _Float16* __restrict__ d, long n) {
    long i = (long)blockIdx.x * blockDim.x + threadIdx.x;
    if (i < n) d[i] = (_Float16)s[i];
}

// ---------------------------------------------------------------------------
// Pack conv weights: wpk[s=kh*3+kw][co][ci_pad] (f16, zero-pad ci>=CI),
// plus fused BN scale/shift:  y = conv*scale + shift; shift folds conv-bias.
// ---------------------------------------------------------------------------
__global__ void pack_conv_kernel(const float* __restrict__ w, const float* __restrict__ bconv,
                                 const float* __restrict__ g, const float* __restrict__ be,
                                 const float* __restrict__ mean, const float* __restrict__ var,
                                 _Float16* __restrict__ wpk, float* __restrict__ scale,
                                 float* __restrict__ shift, int CI, int CIP, int CO) {
    long n = 9L * CO * CIP;
    long i = (long)blockIdx.x * blockDim.x + threadIdx.x;
    if (i < n) {
        int ci = (int)(i % CIP);
        long t = i / CIP;
        int co = (int)(t % CO);
        int s  = (int)(t / CO);
        int kh = s / 3, kw = s % 3;
        float v = 0.f;
        if (ci < CI) v = w[(((long)co * CI + ci) * 3 + kh) * 3 + kw];
        wpk[i] = (_Float16)v;
    }
    if (i < CO) {
        float inv = g[i] * rsqrtf(var[i] + 1e-5f);
        scale[i] = inv;
        shift[i] = (bconv[i] - mean[i]) * inv + be[i];
    }
}

// BN pack for the final 1x1 conv (scale/shift per output channel)
__global__ void pack_bn_kernel(const float* __restrict__ bconv, const float* __restrict__ g,
                               const float* __restrict__ be, const float* __restrict__ mean,
                               const float* __restrict__ var, float* __restrict__ scale,
                               float* __restrict__ shift, int CO) {
    int i = blockIdx.x * blockDim.x + threadIdx.x;
    if (i < CO) {
        float inv = g[i] * rsqrtf(var[i] + 1e-5f);
        scale[i] = inv;
        shift[i] = (bconv[i] - mean[i]) * inv + be[i];
    }
}

// ---------------------------------------------------------------------------
// 3x3 conv as 9 shifted 1x1 WMMA GEMMs, one workgroup per batch element.
//   M = 144 spatial positions (9 waves x 16), N = CO, K = CIP per shift.
//   Input  xin : (B, CI, 144) f16
//   Weights wpk: (9, CO, CIP) f16
//   Output out : (B, CO, 144) f16 = relu(conv*scale + shift)
// ---------------------------------------------------------------------------
template<int CI, int CIP, int CO, int NT>
__global__ __launch_bounds__(288) void conv3x3_kernel(const _Float16* __restrict__ xin,
                                                      const _Float16* __restrict__ wpk,
                                                      const float* __restrict__ scale,
                                                      const float* __restrict__ shift,
                                                      _Float16* __restrict__ out) {
    __shared__ _Float16 ldsX[144 * CIP];
    const int b    = blockIdx.x;
    const int tid  = threadIdx.x;
    const int wave = tid >> 5;
    const int lane = tid & 31;
    const int lh   = lane & 15;
    const int sel  = lane >> 4;

    v8f acc[NT];
#pragma unroll
    for (int nt = 0; nt < NT; ++nt)
#pragma unroll
        for (int r = 0; r < 8; ++r) acc[nt][r] = 0.f;

    for (int s = 0; s < 9; ++s) {
        const int dy = s / 3 - 1, dx = s % 3 - 1;
        __syncthreads();
        // stage shifted, zero-padded input tile: ldsX[p][ci]
        for (int idx = tid; idx < 144 * CIP; idx += 288) {
            int p  = idx % 144;
            int ci = idx / 144;
            int oy = p / 12, ox = p % 12;
            int iy = oy + dy, ix = ox + dx;
            _Float16 v = (_Float16)0.f;
            if (ci < CI && (unsigned)iy < 12u && (unsigned)ix < 12u)
                v = xin[((long)b * CI + ci) * 144 + iy * 12 + ix];
            ldsX[p * CIP + ci] = v;
        }
        __syncthreads();

        const _Float16* arow = ldsX + (wave * 16 + lh) * CIP + sel * 8;
        for (int kb = 0; kb < CIP; kb += 32) {
            v8h alo = *(const v8h*)(arow + kb);
            v8h ahi = *(const v8h*)(arow + kb + 16);
            v16h a  = CAT16(alo, ahi);
#pragma unroll
            for (int nt = 0; nt < NT; ++nt) {
                const _Float16* brow = wpk + ((long)s * CO + nt * 16 + lh) * CIP + kb + sel * 8;
                v8h blo = *(const v8h*)brow;
                v8h bhi = *(const v8h*)(brow + 16);
                v16h bv = CAT16(blo, bhi);
                acc[nt] = __builtin_amdgcn_wmma_f32_16x16x32_f16(
                    false, a, false, bv, (short)0, acc[nt], false, false);
            }
        }
    }

    // epilogue: BN + ReLU, write out[b][co][p] as f16
#pragma unroll
    for (int nt = 0; nt < NT; ++nt) {
        int co = nt * 16 + lh;
        float sc = scale[co], sh = shift[co];
#pragma unroll
        for (int r = 0; r < 8; ++r) {
            int p = wave * 16 + r + sel * 8;
            float v = acc[nt][r] * sc + sh;
            v = fmaxf(v, 0.f);
            out[((long)b * CO + co) * 144 + p] = (_Float16)v;
        }
    }
}

// ---------------------------------------------------------------------------
// Generic tiled WMMA GEMM:  C[M x N] = A(M x K, f16 row-major) * B
//   BKN == false : B supplied as Wt (N x K row-major f16)  -- weights as-is
//   BKN == true  : B supplied as (K x N row-major f16)     -- transposed into LDS
// Epilogue: +biasN[n], +biasM[m], *scaleM[m]+shiftM[m], +res, relu;
// dual f32/f16 outputs with affine (batch,m,n) strides (n-stride == 1).
// Block: 128 threads (4 waves); tile 64x64, K-step 32.
// ---------------------------------------------------------------------------
template<bool BKN>
__global__ __launch_bounds__(128) void gemm_f16_kernel(
    const _Float16* __restrict__ A, long aB, int lda,
    const _Float16* __restrict__ B, long bB, int ldb,
    int M, int N, int K,
    const float* __restrict__ biasN, const float* __restrict__ biasM,
    const float* __restrict__ scaleM, const float* __restrict__ shiftM, int relu,
    const float* __restrict__ res, long rB, int rLd,
    float* __restrict__ out32, long oB32, int oLd32,
    _Float16* __restrict__ out16, long oB16, int oLd16) {
    __shared__ _Float16 ldsA[64 * 40];
    __shared__ _Float16 ldsB[64 * 40];

    const int mbase = blockIdx.x * 64;
    const int nbase = blockIdx.y * 64;
    const int z     = blockIdx.z;
    const _Float16* Ab = A + (long)z * aB;
    const _Float16* Bb = B + (long)z * bB;

    const int tid  = threadIdx.x;
    const int wave = tid >> 5;
    const int lane = tid & 31;
    const int lh   = lane & 15;
    const int sel  = lane >> 4;

    v8f acc[4];
#pragma unroll
    for (int nt = 0; nt < 4; ++nt)
#pragma unroll
        for (int r = 0; r < 8; ++r) acc[nt][r] = 0.f;

    const int ksteps = (K + 31) / 32;
    for (int ks = 0; ks < ksteps; ++ks) {
        const int kb = ks * 32;
        __syncthreads();
        // stage A tile: ldsA[m_local][k_local], zero-filled at edges
        for (int idx = tid; idx < 256; idx += 128) {
            int r = idx >> 2, kg = idx & 3;
            int m = mbase + r, k = kb + kg * 8;
            v8h v = zero_v8h();
            if (m < M && k < K) v = *(const v8h*)(Ab + (long)m * lda + k);
            *(v8h*)(ldsA + r * 40 + kg * 8) = v;
        }
        if (!BKN) {
            // B^T tile: ldsB[n_local][k_local] straight from Wt(N x K)
            for (int idx = tid; idx < 256; idx += 128) {
                int r = idx >> 2, kg = idx & 3;
                int n = nbase + r, k = kb + kg * 8;
                v8h v = zero_v8h();
                if (n < N && k < K) v = *(const v8h*)(Bb + (long)n * ldb + k);
                *(v8h*)(ldsB + r * 40 + kg * 8) = v;
            }
        } else {
            // B given as (K x N): transpose into ldsB[n_local][k_local]
            for (int idx = tid; idx < 256; idx += 128) {
                int kr = idx >> 3, ng = idx & 7;
                int n = nbase + ng * 8, k = kb + kr;
                v8h v = zero_v8h();
                if (k < K && n < N) v = *(const v8h*)(Bb + (long)k * ldb + n);
#pragma unroll
                for (int e = 0; e < 8; ++e) ldsB[(ng * 8 + e) * 40 + kr] = v[e];
            }
        }
        __syncthreads();

        const _Float16* ar = ldsA + (wave * 16 + lh) * 40 + sel * 8;
        v8h alo = *(const v8h*)ar;
        v8h ahi = *(const v8h*)(ar + 16);
        v16h a  = CAT16(alo, ahi);
#pragma unroll
        for (int nt = 0; nt < 4; ++nt) {
            const _Float16* br = ldsB + (nt * 16 + lh) * 40 + sel * 8;
            v8h blo = *(const v8h*)br;
            v8h bhi = *(const v8h*)(br + 16);
            v16h bv = CAT16(blo, bhi);
            acc[nt] = __builtin_amdgcn_wmma_f32_16x16x32_f16(
                false, a, false, bv, (short)0, acc[nt], false, false);
        }
    }

    // epilogue
#pragma unroll
    for (int nt = 0; nt < 4; ++nt) {
        int n = nbase + nt * 16 + lh;
        if (n >= N) continue;
        float bn = biasN ? biasN[n] : 0.f;
#pragma unroll
        for (int r = 0; r < 8; ++r) {
            int m = mbase + wave * 16 + r + sel * 8;
            if (m >= M) continue;
            float v = acc[nt][r] + bn;
            if (biasM)  v += biasM[m];
            if (scaleM) v = v * scaleM[m] + shiftM[m];
            if (res)    v += res[(long)z * rB + (long)m * rLd + n];
            if (relu)   v = fmaxf(v, 0.f);
            if (out32) out32[(long)z * oB32 + (long)m * oLd32 + n] = v;
            if (out16) out16[(long)z * oB16 + (long)m * oLd16 + n] = (_Float16)v;
        }
    }
}

// ---------------------------------------------------------------------------
// Depthwise causal conv1d (kernel 4) + bias + SiLU over the sequence dim.
//   xz : (B*L, 576) f32; xm = cols [0,288)
//   out: xc32 (B*L, 288) f32 and xc16 (B*L, 288) f16
// ---------------------------------------------------------------------------
__global__ void dconv_silu_kernel(const float* __restrict__ xz,
                                  const float* __restrict__ cw,
                                  const float* __restrict__ cb,
                                  float* __restrict__ xc32,
                                  _Float16* __restrict__ xc16) {
    long i = (long)blockIdx.x * blockDim.x + threadIdx.x;
    long n = (long)BATCH * LSEQ * DI;
    if (i >= n) return;
    int d   = (int)(i % DI);
    long bt = i / DI;
    int t   = (int)(bt % LSEQ);
    float acc = cb[d];
#pragma unroll
    for (int j = 0; j < DC; ++j) {
        int ts = t + j - (DC - 1);
        if (ts >= 0) acc += cw[d * DC + j] * xz[(bt + (long)(ts - t)) * (2 * DI) + d];
    }
    float s = siluf(acc);
    xc32[i] = s;
    xc16[i] = (_Float16)s;
}

// ---------------------------------------------------------------------------
// Selective scan: one block per batch, 288 threads (one per DI channel).
// Fuses dtproj + softplus, the 16-state recurrence, and the output gate.
//   dbl : (B*L, 41) f32  [dt 0..8 | B 9..24 | C 25..40]
//   xc  : (B*L, 288) f32 (u, post conv+silu)
//   xz  : (B*L, 576) f32 (z = cols 288..575)
//   yact: (B*L, 288) f16 = (y + u*Dp) * silu(z)
// ---------------------------------------------------------------------------
__global__ __launch_bounds__(288) void scan_kernel(const float* __restrict__ dbl,
                                                   const float* __restrict__ xc,
                                                   const float* __restrict__ xz,
                                                   const float* __restrict__ dtw,
                                                   const float* __restrict__ dtb,
                                                   const float* __restrict__ A_log,
                                                   const float* __restrict__ Dp,
                                                   _Float16* __restrict__ yact) {
    __shared__ float ld[LSEQ * (DR + 2 * DS)];   // 64*41 floats
    const int b = blockIdx.x;
    const int d = threadIdx.x;

    for (int i = threadIdx.x; i < LSEQ * 41; i += 288)
        ld[i] = dbl[(long)b * LSEQ * 41 + i];
    __syncthreads();

    float w[DR];
#pragma unroll
    for (int r = 0; r < DR; ++r) w[r] = dtw[d * DR + r];
    const float bdt = dtb[d];
    float Av[DS];
#pragma unroll
    for (int n = 0; n < DS; ++n) Av[n] = -__expf(A_log[d * DS + n]);
    const float dp = Dp[d];

    float h[DS];
#pragma unroll
    for (int n = 0; n < DS; ++n) h[n] = 0.f;

    for (int t = 0; t < LSEQ; ++t) {
        const float* row = ld + t * 41;
        float dt = bdt;
#pragma unroll
        for (int r = 0; r < DR; ++r) dt += w[r] * row[r];
        float delta = softplusf(dt);
        long bt = (long)b * LSEQ + t;
        float u  = xc[bt * DI + d];
        float zz = xz[bt * (2 * DI) + DI + d];
        float du = delta * u;
        float y = 0.f;
#pragma unroll
        for (int n = 0; n < DS; ++n) {
            h[n] = h[n] * __expf(delta * Av[n]) + du * row[DR + n];
            y += h[n] * row[DR + DS + n];
        }
        yact[bt * DI + d] = (_Float16)((y + u * dp) * siluf(zz));
    }
}

// ---------------------------------------------------------------------------
// Host-side driver
// ---------------------------------------------------------------------------
static inline size_t alup(size_t x) { return (x + 255) & ~(size_t)255; }

extern "C" void kernel_launch(void* const* d_in, const int* in_sizes, int n_in,
                              void* d_out, int out_size, void* d_ws, size_t ws_size,
                              hipStream_t stream) {
    (void)in_sizes; (void)n_in; (void)out_size; (void)ws_size;
    const float* x        = (const float*)d_in[0];
    const float* w1       = (const float*)d_in[1];
    const float* b1       = (const float*)d_in[2];
    const float* g1       = (const float*)d_in[3];
    const float* be1      = (const float*)d_in[4];
    const float* m1       = (const float*)d_in[5];
    const float* v1       = (const float*)d_in[6];
    const float* w2       = (const float*)d_in[7];
    const float* b2       = (const float*)d_in[8];
    const float* g2       = (const float*)d_in[9];
    const float* be2      = (const float*)d_in[10];
    const float* m2       = (const float*)d_in[11];
    const float* v2       = (const float*)d_in[12];
    const float* wc2      = (const float*)d_in[13];
    const float* bc2      = (const float*)d_in[14];
    const float* we       = (const float*)d_in[15];
    const float* bee      = (const float*)d_in[16];
    const float* in_w     = (const float*)d_in[17];
    const float* convd_w  = (const float*)d_in[18];
    const float* convd_b  = (const float*)d_in[19];
    const float* xproj_w  = (const float*)d_in[20];
    const float* dtproj_w = (const float*)d_in[21];
    const float* dtproj_b = (const float*)d_in[22];
    const float* A_log    = (const float*)d_in[23];
    const float* Dp       = (const float*)d_in[24];
    const float* out_w    = (const float*)d_in[25];
    const float* wc3      = (const float*)d_in[26];
    const float* bc3      = (const float*)d_in[27];
    const float* g3       = (const float*)d_in[28];
    const float* be3      = (const float*)d_in[29];
    const float* m3       = (const float*)d_in[30];
    const float* v3       = (const float*)d_in[31];

    char* ws = (char*)d_ws;
    size_t off = 0;
    auto alloc = [&](size_t bytes) -> char* { char* p = ws + off; off += alup(bytes); return p; };

    const long MROWS = (long)BATCH * LSEQ;  // 32768

    _Float16* xh    = (_Float16*)alloc((size_t)BATCH * NB * PP * 2);       // (B,200,144)
    _Float16* wpk1  = (_Float16*)alloc((size_t)9 * 32 * 224 * 2);
    _Float16* wpk2  = (_Float16*)alloc((size_t)9 * 64 * 32 * 2);
    float*    sc1   = (float*)alloc(32 * 4);
    float*    sh1   = (float*)alloc(32 * 4);
    float*    sc2   = (float*)alloc(64 * 4);
    float*    sh2   = (float*)alloc(64 * 4);
    float*    sc3   = (float*)alloc(32 * 4);
    float*    sh3   = (float*)alloc(32 * 4);
    _Float16* h1    = (_Float16*)alloc((size_t)BATCH * 32 * PP * 2);       // (B,32,144)
    _Float16* h2    = (_Float16*)alloc((size_t)BATCH * 64 * PP * 2);       // (B,64,144)
    _Float16* weh   = (_Float16*)alloc((size_t)DM * DM * 2);
    _Float16* h3    = (_Float16*)alloc((size_t)MROWS * DM * 2);            // (B,64,144)
    _Float16* wc2h  = (_Float16*)alloc((size_t)64 * 64 * 2);
    float*    h4f   = (float*)alloc((size_t)MROWS * DM * 4);               // residual
    _Float16* h4h   = (_Float16*)alloc((size_t)MROWS * DM * 2);
    _Float16* inwh  = (_Float16*)alloc((size_t)2 * DI * DM * 2);
    float*    xz32  = (float*)alloc((size_t)MROWS * 2 * DI * 4);
    float*    xc32  = (float*)alloc((size_t)MROWS * DI * 4);
    _Float16* xc16  = (_Float16*)alloc((size_t)MROWS * DI * 2);
    _Float16* xprh  = (_Float16*)alloc((size_t)(DR + 2 * DS) * DI * 2);
    float*    dbl32 = (float*)alloc((size_t)MROWS * (DR + 2 * DS) * 4);
    _Float16* yact  = (_Float16*)alloc((size_t)MROWS * DI * 2);
    _Float16* outwh = (_Float16*)alloc((size_t)DM * DI * 2);
    _Float16* h5    = (_Float16*)alloc((size_t)MROWS * DM * 2);
    _Float16* wc3h  = (_Float16*)alloc((size_t)32 * 64 * 2);

    auto cvt = [&](const float* s, _Float16* d, long n) {
        cvt_f16_kernel<<<dim3((unsigned)((n + 255) / 256)), dim3(256), 0, stream>>>(s, d, n);
    };

    // ---- weight conversion / packing ----
    cvt(x, xh, (long)BATCH * NB * PP);
    cvt(we, weh, (long)DM * DM);
    cvt(in_w, inwh, (long)2 * DI * DM);
    cvt(xproj_w, xprh, (long)(DR + 2 * DS) * DI);
    cvt(out_w, outwh, (long)DM * DI);
    cvt(wc2, wc2h, 64L * 64);
    cvt(wc3, wc3h, 32L * 64);

    {
        long n = 9L * 32 * 224;
        pack_conv_kernel<<<dim3((unsigned)((n + 255) / 256)), dim3(256), 0, stream>>>(
            w1, b1, g1, be1, m1, v1, wpk1, sc1, sh1, NB, 224, 32);
    }
    {
        long n = 9L * 64 * 32;
        pack_conv_kernel<<<dim3((unsigned)((n + 255) / 256)), dim3(256), 0, stream>>>(
            w2, b2, g2, be2, m2, v2, wpk2, sc2, sh2, 32, 32, 64);
    }
    pack_bn_kernel<<<dim3(1), dim3(32), 0, stream>>>(bc3, g3, be3, m3, v3, sc3, sh3, 32);

    // ---- conv1: (B,200,144) -> (B,32,144), BN+ReLU ----
    conv3x3_kernel<NB, 224, 32, 2><<<dim3(BATCH), dim3(288), 0, stream>>>(xh, wpk1, sc1, sh1, h1);
    // ---- conv2: (B,32,144) -> (B,64,144), BN+ReLU ----
    conv3x3_kernel<32, 32, 64, 4><<<dim3(BATCH), dim3(288), 0, stream>>>(h1, wpk2, sc2, sh2, h2);

    // ---- we linear: h3 = h2 @ we.T + bee   (M=32768, N=144, K=144) ----
    gemm_f16_kernel<false><<<dim3(512, 3, 1), dim3(128), 0, stream>>>(
        h2, 0L, PP, weh, 0L, DM, (int)MROWS, DM, PP,
        bee, nullptr, nullptr, nullptr, 0,
        nullptr, 0L, 0, nullptr, 0L, 0, h3, 0L, DM);

    // ---- wc2 1x1 conv (batched): h4[b,co,dm] = wc2[co,l] @ h3[b,l,dm] + bc2 ----
    gemm_f16_kernel<true><<<dim3(1, 3, BATCH), dim3(128), 0, stream>>>(
        wc2h, 0L, 64, h3, (long)LSEQ * DM, DM, LSEQ, DM, LSEQ,
        nullptr, bc2, nullptr, nullptr, 0,
        nullptr, 0L, 0, h4f, (long)LSEQ * DM, DM, h4h, (long)LSEQ * DM, DM);

    // ---- in_w: xz = h4 @ in_w.T   (M=32768, N=576, K=144) ----
    gemm_f16_kernel<false><<<dim3(512, 9, 1), dim3(128), 0, stream>>>(
        h4h, 0L, DM, inwh, 0L, DM, (int)MROWS, 2 * DI, DM,
        nullptr, nullptr, nullptr, nullptr, 0,
        nullptr, 0L, 0, xz32, 0L, 2 * DI, nullptr, 0L, 0);

    // ---- depthwise causal conv + SiLU ----
    {
        long n = MROWS * DI;
        dconv_silu_kernel<<<dim3((unsigned)((n + 255) / 256)), dim3(256), 0, stream>>>(
            xz32, convd_w, convd_b, xc32, xc16);
    }

    // ---- xproj: dbl = xc @ xproj_w.T   (M=32768, N=41, K=288) ----
    gemm_f16_kernel<false><<<dim3(512, 1, 1), dim3(128), 0, stream>>>(
        xc16, 0L, DI, xprh, 0L, DI, (int)MROWS, DR + 2 * DS, DI,
        nullptr, nullptr, nullptr, nullptr, 0,
        nullptr, 0L, 0, dbl32, 0L, DR + 2 * DS, nullptr, 0L, 0);

    // ---- selective scan (fused dtproj+softplus, recurrence, gate) ----
    scan_kernel<<<dim3(BATCH), dim3(288), 0, stream>>>(
        dbl32, xc32, xz32, dtproj_w, dtproj_b, A_log, Dp, yact);

    // ---- out proj + residual: h5 = yact @ out_w.T + h4   (M=32768, N=144, K=288) ----
    gemm_f16_kernel<false><<<dim3(512, 3, 1), dim3(128), 0, stream>>>(
        yact, 0L, DI, outwh, 0L, DI, (int)MROWS, DM, DI,
        nullptr, nullptr, nullptr, nullptr, 0,
        h4f, 0L, DM, nullptr, 0L, 0, h5, 0L, DM);

    // ---- wc3 1x1 conv + BN + ReLU -> d_out (B, 32*144) ----
    gemm_f16_kernel<true><<<dim3(1, 3, BATCH), dim3(128), 0, stream>>>(
        wc3h, 0L, 64, h5, (long)LSEQ * DM, DM, 32, DM, LSEQ,
        nullptr, nullptr, sc3, sh3, 1,
        nullptr, 0L, 0, (float*)d_out, (long)32 * DM, DM, nullptr, 0L, 0);
}
